// MPNN_91285234909505
// MI455X (gfx1250) — matrix operand order repaired and checked by reading, not compiled, not run on hardware
//
#include <hip/hip_runtime.h>
#include <hip/hip_bf16.h>

typedef __attribute__((ext_vector_type(16))) _Float16 v16h;
typedef __attribute__((ext_vector_type(8)))  _Float16 v8h;
typedef __attribute__((ext_vector_type(8)))  float    v8f;

#define N_NODES 50000
#define N_EDGES 800000
#define D_N   64
#define D_E   64
#define HID   128
#define D_OUT 64
#define XE_DIM 192          // 2*D_N + D_E
#define XN_DIM 128          // D_OUT + D_N
#define EB 128              // edges per block
#define NB 128              // nodes per block
#define NWAVES 8            // 256 threads = 8 wave32

// ---------------------------------------------------------------------------
// WMMA operand loaders (CDNA5 wave32 layouts, 16-bit data).
//
// A (16x32, MxK), row = lane&15:
//   lanes 0-15 : halves 0..7 = K+0..7,  halves 8..15 = K+16..23
//   lanes 16-31: halves 0..7 = K+8..15, halves 8..15 = K+24..31
__device__ __forceinline__ v16h ld_a_tile(const _Float16* p) {
  v8h lo = *(const v8h*)(p);
  v8h hi = *(const v8h*)(p + 16);
  v16h r;
#pragma unroll
  for (int i = 0; i < 8; ++i) { r[i] = lo[i]; r[i + 8] = hi[i]; }
  return r;
}

// B (32x16, KxN), col = lane&15 (same K order as A, from the
//   [N][K]-transposed weight array):
//   lanes 0-15 : halves 0..7 = K+0..7,  halves 8..15 = K+16..23
//   lanes 16-31: halves 0..7 = K+8..15, halves 8..15 = K+24..31
__device__ __forceinline__ v16h ld_b_tile(const _Float16* p) {
  v8h lo = *(const v8h*)(p);
  v8h hi = *(const v8h*)(p + 16);
  v16h r;
#pragma unroll
  for (int i = 0; i < 8; ++i) { r[i] = lo[i]; r[i + 8] = hi[i]; }
  return r;
}

// Pack two f32 -> two f16 -> one 32-bit LDS store.
__device__ __forceinline__ void st_pk_f16(_Float16* p, float2 v) {
  union { _Float16 h[2]; unsigned u; } x;
  x.h[0] = (_Float16)v.x;
  x.h[1] = (_Float16)v.y;
  *(unsigned*)p = x.u;
}

// ---------------------------------------------------------------------------
// Tiled GEMM fragment: one wave, one N-tile (n0), MTILES M-tiles starting at
// row mbase, K = KDIM.  B tile is loaded ONCE per k-step and reused across all
// M-tiles (register reuse -> minimal WGP$/L2 weight traffic).
// A: LDS, row-major [M][KDIM] f16.  Wt: global, row-major [N][KDIM] f16.
template <int KDIM, int MTILES>
__device__ __forceinline__ void gemm_tiles(const _Float16* A,
                                           const _Float16* __restrict__ Wt,
                                           const float* __restrict__ bias,
                                           int n0, int mbase, int lane,
                                           v8f* acc) {
  const int lrow = lane & 15;
  const int lk   = (lane >> 4) << 3;   // A k-offset (0/8), C/D m-offset
  const int bk   = (lane >> 4) << 3;   // B k-offset (0/8)
  const float b = bias[n0 + lrow];
#pragma unroll
  for (int mt = 0; mt < MTILES; ++mt)
#pragma unroll
    for (int r = 0; r < 8; ++r) acc[mt][r] = b;
#pragma unroll
  for (int k0 = 0; k0 < KDIM; k0 += 32) {
    const v16h bm = ld_b_tile(Wt + (size_t)(n0 + lrow) * KDIM + k0 + bk);
#pragma unroll
    for (int mt = 0; mt < MTILES; ++mt) {
      v16h a = ld_a_tile(A + (mbase + mt * 16 + lrow) * KDIM + k0 + lk);
      acc[mt] = __builtin_amdgcn_wmma_f32_16x16x32_f16(false, a, false, bm,
                                                       (short)0, acc[mt],
                                                       false, false);
    }
  }
}

// ---------------------------------------------------------------------------
// Prep: transpose f32 weight [K][N] -> f16 [N][K] (contiguous K rows for B).
__global__ void wt_transpose_f16(const float* __restrict__ W,
                                 _Float16* __restrict__ Wt, int K, int N) {
  int i = blockIdx.x * blockDim.x + threadIdx.x;
  if (i >= K * N) return;
  int n = i / K;
  int k = i - n * K;
  Wt[(size_t)n * K + k] = (_Float16)W[(size_t)k * N + n];
}

__global__ void zero_f32(float* __restrict__ p, int n) {
  int i = blockIdx.x * blockDim.x + threadIdx.x;
  if (i < n) p[i] = 0.0f;
}

// ---------------------------------------------------------------------------
// Edge kernel: x_e = [nf[src]|nf[dst]|ef]; e_out = MLP_e(x_e);
// gate = sigmoid(MLP_a(x_e)); updated_ef = gate*e_out; atomic scatter to agg.
__global__ __launch_bounds__(256) void mpnn_edge_kernel(
    const float* __restrict__ nf, const float* __restrict__ ef,
    const int* __restrict__ src, const int* __restrict__ dst,
    const _Float16* __restrict__ We1t, const float* __restrict__ be1,
    const _Float16* __restrict__ We2t, const float* __restrict__ be2,
    const _Float16* __restrict__ Wa1t, const float* __restrict__ ba1,
    const _Float16* __restrict__ Wa2t, const float* __restrict__ ba2,
    float* __restrict__ out_ef, float* __restrict__ agg)
{
  __shared__ alignas(16) _Float16 sX[EB * XE_DIM];   // 48 KB
  __shared__ alignas(16) _Float16 sH[EB * HID];      // 32 KB
  __shared__ alignas(16) float    sO[EB * D_OUT];    // 32 KB
  __shared__ int sDst[EB];

  const int tid  = threadIdx.x;
  const int wave = tid >> 5;
  const int lane = tid & 31;
  const int lrow = lane & 15;
  const int lk   = (lane >> 4) << 3;
  const int e0   = blockIdx.x * EB;

  // ---- gather x_e rows into LDS as f16 (one wave per row, 2 elems/lane) ----
  for (int er = wave; er < EB; er += NWAVES) {
    const int e = e0 + er;
    const int s = src[e];
    const int d = dst[e];
    if (lane == 0) sDst[er] = d;
    const float2 vs = *(const float2*)(nf + (size_t)s * D_N + lane * 2);
    const float2 vd = *(const float2*)(nf + (size_t)d * D_N + lane * 2);
    const float2 ve = *(const float2*)(ef + (size_t)e * D_E + lane * 2);
    _Float16* row = sX + er * XE_DIM;
    st_pk_f16(row + lane * 2, vs);
    st_pk_f16(row + 64 + lane * 2, vd);
    st_pk_f16(row + 128 + lane * 2, ve);
  }
  __syncthreads();

  // ===== MLP-e layer 1: (128x192)@(192x128)+be1, ReLU -> sH =====
  // wave w owns N-tile n0 = w*16, all 8 M-tiles (B reused 8x from registers).
  {
    v8f acc[8];
    gemm_tiles<XE_DIM, 8>(sX, We1t, be1, wave * 16, 0, lane, acc);
#pragma unroll
    for (int mt = 0; mt < 8; ++mt)
#pragma unroll
      for (int r = 0; r < 8; ++r) {
        float v = fmaxf(acc[mt][r], 0.0f);
        sH[(mt * 16 + r + lk) * HID + wave * 16 + lrow] = (_Float16)v;
      }
  }
  __syncthreads();

  // ===== MLP-e layer 2: (128x128)@(128x64)+be2 -> sO (f32) =====
  // wave w owns n0=(w&3)*16, M-tiles [(w>>2)*64 .. +63] (B reused 4x).
  {
    const int n0 = (wave & 3) * 16;
    const int mbase = (wave >> 2) * 64;
    v8f acc[4];
    gemm_tiles<HID, 4>(sH, We2t, be2, n0, mbase, lane, acc);
#pragma unroll
    for (int mt = 0; mt < 4; ++mt)
#pragma unroll
      for (int r = 0; r < 8; ++r)
        sO[(mbase + mt * 16 + r + lk) * D_OUT + n0 + lrow] = acc[mt][r];
  }
  __syncthreads();

  // ===== MLP-a layer 1 (overwrite sH) =====
  {
    v8f acc[8];
    gemm_tiles<XE_DIM, 8>(sX, Wa1t, ba1, wave * 16, 0, lane, acc);
#pragma unroll
    for (int mt = 0; mt < 8; ++mt)
#pragma unroll
      for (int r = 0; r < 8; ++r) {
        float v = fmaxf(acc[mt][r], 0.0f);
        sH[(mt * 16 + r + lk) * HID + wave * 16 + lrow] = (_Float16)v;
      }
  }
  __syncthreads();

  // ===== MLP-a layer 2 + sigmoid, fuse with sO, store + scatter-add =====
  {
    const int n0 = (wave & 3) * 16;
    const int mbase = (wave >> 2) * 64;
    v8f acc[4];
    gemm_tiles<HID, 4>(sH, Wa2t, ba2, n0, mbase, lane, acc);
#pragma unroll
    for (int mt = 0; mt < 4; ++mt)
#pragma unroll
      for (int r = 0; r < 8; ++r) {
        const int m = mbase + mt * 16 + r + lk;
        const int n = n0 + lrow;
        const float g = 1.0f / (1.0f + __expf(-acc[mt][r]));
        const float v = g * sO[m * D_OUT + n];
        out_ef[(size_t)(e0 + m) * D_OUT + n] = v;
        atomicAdd(&agg[(size_t)sDst[m] * D_OUT + n], v);   // L2-resident
      }
  }
}

// ---------------------------------------------------------------------------
// Node kernel: x_n = [agg|nf]; updated_nf = relu(x_n@Wn1+bn1)@Wn2+bn2
__global__ __launch_bounds__(256) void mpnn_node_kernel(
    const float* __restrict__ nf, const float* __restrict__ agg,
    const _Float16* __restrict__ Wn1t, const float* __restrict__ bn1,
    const _Float16* __restrict__ Wn2t, const float* __restrict__ bn2,
    float* __restrict__ out_nf)
{
  __shared__ alignas(16) _Float16 sX[NB * XN_DIM];   // 32 KB
  __shared__ alignas(16) _Float16 sH[NB * HID];      // 32 KB

  const int tid  = threadIdx.x;
  const int wave = tid >> 5;
  const int lane = tid & 31;
  const int lrow = lane & 15;
  const int lk   = (lane >> 4) << 3;
  const int v0   = blockIdx.x * NB;

  for (int nr = wave; nr < NB; nr += NWAVES) {
    const int node = v0 + nr;
    _Float16* row = sX + nr * XN_DIM;
    if (node < N_NODES) {
      const float2 va = *(const float2*)(agg + (size_t)node * D_OUT + lane * 2);
      const float2 vn = *(const float2*)(nf + (size_t)node * D_N + lane * 2);
      st_pk_f16(row + lane * 2, va);
      st_pk_f16(row + 64 + lane * 2, vn);
    } else {
      st_pk_f16(row + lane * 2, make_float2(0.f, 0.f));
      st_pk_f16(row + 64 + lane * 2, make_float2(0.f, 0.f));
    }
  }
  __syncthreads();

  // layer 1: (128x128)@(128x128)+bn1, ReLU -> sH
  {
    v8f acc[8];
    gemm_tiles<XN_DIM, 8>(sX, Wn1t, bn1, wave * 16, 0, lane, acc);
#pragma unroll
    for (int mt = 0; mt < 8; ++mt)
#pragma unroll
      for (int r = 0; r < 8; ++r) {
        float v = fmaxf(acc[mt][r], 0.0f);
        sH[(mt * 16 + r + lk) * HID + wave * 16 + lrow] = (_Float16)v;
      }
  }
  __syncthreads();

  // layer 2: (128x128)@(128x64)+bn2 -> out_nf (guarded)
  {
    const int n0 = (wave & 3) * 16;
    const int mbase = (wave >> 2) * 64;
    v8f acc[4];
    gemm_tiles<HID, 4>(sH, Wn2t, bn2, n0, mbase, lane, acc);
#pragma unroll
    for (int mt = 0; mt < 4; ++mt)
#pragma unroll
      for (int r = 0; r < 8; ++r) {
        const int node = v0 + mbase + mt * 16 + r + lk;
        if (node < N_NODES)
          out_nf[(size_t)node * D_OUT + n0 + lrow] = acc[mt][r];
      }
  }
}

// ---------------------------------------------------------------------------
extern "C" void kernel_launch(void* const* d_in, const int* in_sizes, int n_in,
                              void* d_out, int out_size, void* d_ws, size_t ws_size,
                              hipStream_t stream) {
  (void)in_sizes; (void)n_in; (void)out_size; (void)ws_size;

  const float* nf  = (const float*)d_in[0];
  const float* ef  = (const float*)d_in[1];
  const int*   src = (const int*)d_in[2];
  const int*   dst = (const int*)d_in[3];
  const float* We1 = (const float*)d_in[4];
  const float* be1 = (const float*)d_in[5];
  const float* We2 = (const float*)d_in[6];
  const float* be2 = (const float*)d_in[7];
  const float* Wa1 = (const float*)d_in[8];
  const float* ba1 = (const float*)d_in[9];
  const float* Wa2 = (const float*)d_in[10];
  const float* ba2 = (const float*)d_in[11];
  const float* Wn1 = (const float*)d_in[12];
  const float* bn1 = (const float*)d_in[13];
  const float* Wn2 = (const float*)d_in[14];
  const float* bn2 = (const float*)d_in[15];

  // workspace layout: transposed f16 weights, then f32 agg buffer
  char* ws = (char*)d_ws;
  _Float16* We1t = (_Float16*)ws;          // [128][192]
  _Float16* Wa1t = We1t + 128 * 192;       // [128][192]
  _Float16* We2t = Wa1t + 128 * 192;       // [64][128]
  _Float16* Wa2t = We2t + 64 * 128;        // [64][128]
  _Float16* Wn1t = Wa2t + 64 * 128;        // [128][128]
  _Float16* Wn2t = Wn1t + 128 * 128;       // [64][128]
  float* agg = (float*)(ws + 180224);      // 16B-aligned, [N_NODES][64] f32

  const int T = 256;
  wt_transpose_f16<<<(192 * 128 + T - 1) / T, T, 0, stream>>>(We1, We1t, 192, 128);
  wt_transpose_f16<<<(192 * 128 + T - 1) / T, T, 0, stream>>>(Wa1, Wa1t, 192, 128);
  wt_transpose_f16<<<(128 * 64 + T - 1) / T, T, 0, stream>>>(We2, We2t, 128, 64);
  wt_transpose_f16<<<(128 * 64 + T - 1) / T, T, 0, stream>>>(Wa2, Wa2t, 128, 64);
  wt_transpose_f16<<<(128 * 128 + T - 1) / T, T, 0, stream>>>(Wn1, Wn1t, 128, 128);
  wt_transpose_f16<<<(128 * 64 + T - 1) / T, T, 0, stream>>>(Wn2, Wn2t, 128, 64);
  zero_f32<<<(N_NODES * D_OUT + T - 1) / T, T, 0, stream>>>(agg, N_NODES * D_OUT);

  float* out_nf = (float*)d_out;                       // [N_NODES][64]
  float* out_ef = out_nf + (size_t)N_NODES * D_OUT;    // [N_EDGES][64]

  mpnn_edge_kernel<<<N_EDGES / EB, 256, 0, stream>>>(
      nf, ef, src, dst, We1t, be1, We2t, be2, Wa1t, ba1, Wa2t, ba2,
      out_ef, agg);
  mpnn_node_kernel<<<(N_NODES + NB - 1) / NB, 256, 0, stream>>>(
      nf, agg, Wn1t, bn1, Wn2t, bn2, out_nf);
}
